// HashMLP_80530636800408
// MI455X (gfx1250) — matrix operand order repaired and checked by reading, not compiled, not run on hardware
//
#include <hip/hip_runtime.h>

typedef _Float16     v16h __attribute__((ext_vector_type(16)));
typedef float        v8f  __attribute__((ext_vector_type(8)));
typedef unsigned int v8u  __attribute__((ext_vector_type(8)));

#define LVLS   16
#define TSIZE  (1u << 19)
#define NPTS   (1 << 20)
#define HSTR   136            // padded f16 row stride (68 dwords -> bank-friendly)
#define WAVES  8

// ws layout (u32 units): P0[2048] P1[8192] P2[8192] P3[1024] RES[16]
#define P0_OFF 0
#define P1_OFF 2048
#define P2_OFF 10240
#define P3_OFF 18432
#define RS_OFF 19456
#define PREP_N 19472

__device__ __forceinline__ unsigned short f2h(float f) {
    return __builtin_bit_cast(unsigned short, (_Float16)f);
}

__device__ __forceinline__ float fast_tanh(float z) {
#if __has_builtin(__builtin_amdgcn_tanhf)
    return __builtin_amdgcn_tanhf(z);            // v_tanh_f32 (gfx1250 TRANS op)
#elif __has_builtin(__builtin_amdgcn_tanh_f32)
    return __builtin_amdgcn_tanh_f32(z);
#else
    return tanhf(z);
#endif
}

// tanh-form GELU: branchless, ~6 VALU (1 TRANS) vs ~40 branchy ops for erff
__device__ __forceinline__ float gelu_fast(float x) {
    float inner = 0.7978845608028654f * x * __builtin_fmaf(0.044715f, x * x, 1.f);
    float h = 0.5f * x;
    return __builtin_fmaf(h, fast_tanh(inner), h);
}

// ---------------------------------------------------------------------------
// Prep: (a) pack W0..W3 (f32 [fan_in][fan_out]) into f16 WMMA B-fragment order
//           P[nt][kt][lane][v] = pack(W[k0][n], W[k0+1][n]),
//           n = nt*16 + lane%16, k0 = kt*32 + (lane/16)*16 + 2v
//       (b) compute the 16 per-level resolutions in float64, like the reference
// ---------------------------------------------------------------------------
__global__ void prep_kernel(const float* __restrict__ W0, const float* __restrict__ W1,
                            const float* __restrict__ W2, const float* __restrict__ W3,
                            unsigned* __restrict__ ws) {
    int e = blockIdx.x * blockDim.x + threadIdx.x;
    if (e >= PREP_N) return;
    if (e >= RS_OFF) {                       // RES table
        int l = e - RS_OFF;
        double lg = (log(2048.0) - log(16.0)) / 15.0;
        ((float*)ws)[e] = (float)floor(16.0 * exp((double)l * lg));
        return;
    }
    const float* W; unsigned* P; int Nf, ktiles;
    if (e < P1_OFF)      { W = W0; P = ws + P0_OFF;              Nf = 128; ktiles = 1; }
    else if (e < P2_OFF) { e -= P1_OFF; W = W1; P = ws + P1_OFF; Nf = 128; ktiles = 4; }
    else if (e < P3_OFF) { e -= P2_OFF; W = W2; P = ws + P2_OFF; Nf = 128; ktiles = 4; }
    else                 { e -= P3_OFF; W = W3; P = ws + P3_OFF; Nf = 3;   ktiles = 4; }
    int v    = e & 7;
    int lane = (e >> 3) & 31;
    int kt   = (e >> 8) % ktiles;
    int nt   = (e >> 8) / ktiles;
    int n    = nt * 16 + (lane & 15);
    int k0   = kt * 32 + (lane >> 4) * 16 + 2 * v;
    float lo = (n < Nf) ? W[k0 * Nf + n]       : 0.f;
    float hi = (n < Nf) ? W[(k0 + 1) * Nf + n] : 0.f;
    P[e] = (unsigned)f2h(lo) | ((unsigned)f2h(hi) << 16);
}

// A fragment (16x32 f16) from padded LDS row-major tile, element s holds k = kh*16 + s (contiguous, same as B packing)
__device__ __forceinline__ v16h loadA(const _Float16* __restrict__ row, int kh) {
    v8u u;
#pragma unroll
    for (int v = 0; v < 8; ++v) {
        int k0 = kh * 16 + 2 * v;
        u[v] = *(const unsigned*)(row + k0);
    }
    return __builtin_bit_cast(v16h, u);
}

__device__ __forceinline__ v16h loadB(const unsigned* __restrict__ p) {
    v8u u = *(const v8u*)p;                  // 32B-aligned vector load
    return __builtin_bit_cast(v16h, u);
}

__global__ __launch_bounds__(256) void hashmlp_kernel(
    const float* __restrict__ x, const float* __restrict__ tables,
    const unsigned* __restrict__ ws,
    const float* __restrict__ b0, const float* __restrict__ b1,
    const float* __restrict__ b2, const float* __restrict__ b3,
    float* __restrict__ out) {
    __shared__ _Float16 hbuf[WAVES][16][HSTR];   // 34,816 B
    const unsigned* __restrict__ P0 = ws + P0_OFF;
    const unsigned* __restrict__ P1 = ws + P1_OFF;
    const unsigned* __restrict__ P2 = ws + P2_OFF;
    const unsigned* __restrict__ P3 = ws + P3_OFF;
    const float*    __restrict__ rs = (const float*)(ws + RS_OFF);

    const int tid  = threadIdx.x;
    const int w    = tid >> 5;
    const int lane = tid & 31;
    const int m16  = lane & 15;
    const int kh   = lane >> 4;
    const int ptBase = blockIdx.x * (WAVES * 16) + w * 16;
    _Float16 (* __restrict__ hb)[HSTR] = hbuf[w];

    // ---------------- hash encoding: lane handles point m16, levels [8*kh, 8*kh+8)
    {
        const float x0 = x[2 * (ptBase + m16)];
        const float x1 = x[2 * (ptBase + m16) + 1];
        const float4 ra = ((const float4*)(rs + kh * 8))[0];
        const float4 rb = ((const float4*)(rs + kh * 8))[1];
        const float rv[8] = {ra.x, ra.y, ra.z, ra.w, rb.x, rb.y, rb.z, rb.w};
#pragma unroll
        for (int i = 0; i < 8; ++i) {
            const int l = kh * 8 + i;
            const float r = rv[i];
            const float px = x0 * r, py = x1 * r;
            const float fx = floorf(px), fy = floorf(py);
            const float wx = px - fx, wy = py - fy;
            const int ix = (int)fx, iy = (int)fy;
            const float2* __restrict__ tab = (const float2*)tables + (size_t)l * TSIZE;
            float f0 = 0.f, f1 = 0.f;
#pragma unroll
            for (int c = 0; c < 4; ++c) {
                const int dx = c >> 1, dy = c & 1;
                const unsigned hx = (unsigned)(ix + dx);
                const unsigned hy = (unsigned)(iy + dy);
                const unsigned idx = (hx ^ (hy * 2654435761u)) & (TSIZE - 1u);
                const float2 t = tab[idx];
                const float wt = (dx ? wx : 1.f - wx) * (dy ? wy : 1.f - wy);
                f0 += t.x * wt; f1 += t.y * wt;
            }
            *(unsigned*)&hb[m16][2 * l] = (unsigned)f2h(f0) | ((unsigned)f2h(f1) << 16);
        }
    }
    // per-wave LDS tile; same-wave DS ops are in order -> no barrier needed

    // ---------------- layer 0: 32 -> 128, GELU (8 WMMAs, nt loop rolled)
    {
        const v16h a0 = loadA(&hb[m16][0], kh);
#pragma unroll 1
        for (int nt = 0; nt < 8; ++nt) {
            const float bv = b0[nt * 16 + m16];
            v8f c = {bv, bv, bv, bv, bv, bv, bv, bv};
            const v16h b = loadB(P0 + (nt * 32 + lane) * 8);
            c = __builtin_amdgcn_wmma_f32_16x16x32_f16(false, a0, false, b,
                                                       (short)0, c, false, false);
#pragma unroll
            for (int v = 0; v < 8; ++v)
                hb[v + 8 * kh][nt * 16 + m16] = (_Float16)gelu_fast(c[v]);
        }
    }

    // ---------------- layers 1 & 2: 128 -> 128, GELU (32 WMMAs each)
#pragma unroll 1
    for (int layer = 0; layer < 2; ++layer) {
        const unsigned* __restrict__ P  = layer ? P2 : P1;
        const float*    __restrict__ bb = layer ? b2 : b1;
        v16h a[4];
#pragma unroll
        for (int kt = 0; kt < 4; ++kt) a[kt] = loadA(&hb[m16][kt * 32], kh);
#pragma unroll 1
        for (int nt = 0; nt < 8; ++nt) {
            const float bv = bb[nt * 16 + m16];
            v8f c = {bv, bv, bv, bv, bv, bv, bv, bv};
#pragma unroll
            for (int kt = 0; kt < 4; ++kt) {
                const v16h b = loadB(P + ((nt * 4 + kt) * 32 + lane) * 8);
                c = __builtin_amdgcn_wmma_f32_16x16x32_f16(false, a[kt], false, b,
                                                           (short)0, c, false, false);
            }
#pragma unroll
            for (int v = 0; v < 8; ++v)
                hb[v + 8 * kh][nt * 16 + m16] = (_Float16)gelu_fast(c[v]);
        }
    }

    // ---------------- layer 3: 128 -> 3 (padded to one 16-wide N tile, 4 WMMAs)
    {
        v16h a[4];
#pragma unroll
        for (int kt = 0; kt < 4; ++kt) a[kt] = loadA(&hb[m16][kt * 32], kh);
        const float bv = (m16 < 3) ? b3[m16] : 0.f;
        v8f c = {bv, bv, bv, bv, bv, bv, bv, bv};
#pragma unroll
        for (int kt = 0; kt < 4; ++kt) {
            const v16h b = loadB(P3 + (kt * 32 + lane) * 8);
            c = __builtin_amdgcn_wmma_f32_16x16x32_f16(false, a[kt], false, b,
                                                       (short)0, c, false, false);
        }
        if (m16 < 3) {
#pragma unroll
            for (int v = 0; v < 8; ++v)
                out[(size_t)(ptBase + v + 8 * kh) * 3 + m16] = c[v];
        }
    }
}

extern "C" void kernel_launch(void* const* d_in, const int* in_sizes, int n_in,
                              void* d_out, int out_size, void* d_ws, size_t ws_size,
                              hipStream_t stream) {
    const float* x      = (const float*)d_in[0];
    const float* tables = (const float*)d_in[1];
    const float* W0 = (const float*)d_in[2];
    const float* b0 = (const float*)d_in[3];
    const float* W1 = (const float*)d_in[4];
    const float* b1 = (const float*)d_in[5];
    const float* W2 = (const float*)d_in[6];
    const float* b2 = (const float*)d_in[7];
    const float* W3 = (const float*)d_in[8];
    const float* b3 = (const float*)d_in[9];
    unsigned* ws = (unsigned*)d_ws;

    prep_kernel<<<(PREP_N + 255) / 256, 256, 0, stream>>>(W0, W1, W2, W3, ws);
    hashmlp_kernel<<<NPTS / (WAVES * 16), WAVES * 32, 0, stream>>>(
        x, tables, ws, b0, b1, b2, b3, (float*)d_out);
}
